// VWE_69320772157716
// MI455X (gfx1250) — compile-verified, matrix operand order checked by reading, not run on hardware
//
#include <hip/hip_runtime.h>
#include <hip/hip_bf16.h>
#include <math.h>

typedef __attribute__((ext_vector_type(16))) __bf16 v16bf;
typedef __attribute__((ext_vector_type(8)))  float  v8f;

#define NB   16
#define NC   2048
#define NK   256
#define NHW  1024           // 32*32
#define KSTEPS (NC / 32)    // 64 k-steps of 32 (bf16 WMMA K)
#define NTILE  (NK / 16)    // 16 K-tiles of 16
#define MP    32            // pixels per workgroup (2 WMMA M-subtiles)

__device__ __forceinline__ unsigned short f2bf(float f) {
  unsigned int u = __float_as_uint(f);
  u += 0x7FFFu + ((u >> 16) & 1u);            // round-to-nearest-even
  return (unsigned short)(u >> 16);
}

// ISA 16-bit 16x32 fragment layout (A; B mirrors with row=N):
// element e (0..15) of v16bf, row, k(0..31)  ->  (lane, e)
__device__ __forceinline__ void frag_pos(int row, int k, int& lane, int& e) {
  int khalf = k & 15;
  int hi = k >> 4;                             // 0: K 0-15, 1: K 16-31
  if (khalf < 8) { lane = row;      e = hi * 8 + khalf; }
  else           { lane = row + 16; e = hi * 8 + (khalf - 8); }
}

// ---------------- kernel 0: zero hist + y_word regions of d_out -------------
__global__ __launch_bounds__(256) void k_zero(float* __restrict__ p, int n) {
  int i = blockIdx.x * 256 + threadIdx.x;
  if (i < n) p[i] = 0.0f;
}

// ---------------- kernel 1: centroid inverse norms --------------------------
__global__ __launch_bounds__(256) void k_wnorm(const float* __restrict__ w,
                                               float* __restrict__ w_invn) {
  __shared__ float red[256];
  int k = blockIdx.x;
  float ss = 0.f;
  for (int c = threadIdx.x; c < NC; c += 256) {
    float v = w[k * NC + c];
    ss += v * v;
  }
  red[threadIdx.x] = ss;
  __syncthreads();
  for (int s = 128; s > 0; s >>= 1) {
    if (threadIdx.x < s) red[threadIdx.x] += red[threadIdx.x + s];
    __syncthreads();
  }
  if (threadIdx.x == 0)
    w_invn[k] = 1.0f / fmaxf(sqrtf(red[0]), 1e-12f);
}

// ---------------- kernel 2: normalized bf16 weights in B-fragment order -----
// wfrag[tile][step][lane][e] , 16*64*32*16 bf16 = 1 MB (L2-resident)
__global__ __launch_bounds__(256) void k_wfrag(const float* __restrict__ w,
                                               const float* __restrict__ w_invn,
                                               unsigned short* __restrict__ wfrag) {
  int idx  = blockIdx.x * 256 + threadIdx.x;   // 0 .. 524287
  int tile = idx >> 15;                        // / 32768
  int rem  = idx & 32767;
  int step = rem >> 9;                         // / 512
  int lane = (rem >> 4) & 31;
  int e    = rem & 15;
  int n    = tile * 16 + (lane & 15);
  int k    = ((e < 8) ? 0 : 16) + ((lane < 16) ? (e & 7) : (e & 7) + 8);
  int c    = step * 32 + k;
  wfrag[idx] = f2bf(w[n * NC + c] * w_invn[n]);
}

// ---------------- kernel 3: per-pixel inverse norms of x --------------------
__global__ __launch_bounds__(256) void k_xnorm(const float* __restrict__ x,
                                               float* __restrict__ x_invn) {
  int p  = blockIdx.x * 256 + threadIdx.x;     // pixel id, 0..16383
  int b  = p >> 10;
  int hw = p & 1023;
  const float* xp = x + (size_t)b * NC * NHW + hw;
  float ss = 0.f;
  for (int c = 0; c < NC; ++c) {
    float v = xp[(size_t)c * NHW];
    ss += v * v;
  }
  x_invn[p] = 1.0f / fmaxf(sqrtf(ss), 1e-12f);
}

// ---------------- kernel 4: fused GEMM + softmax + argmax + hist ------------
// grid 512 (16 images x 32 pixel-tiles of 32), block 512 (16 wave32 waves).
// Each wave owns one 16-wide K tile and multiplies ONE B fragment against
// TWO A sub-tiles per k-step -> halves B/L2 traffic vs 16-pixel tiles.
__global__ __launch_bounds__(512) void k_main(const float* __restrict__ x,
                                              const unsigned short* __restrict__ wfrag,
                                              const float* __restrict__ x_invn,
                                              float* __restrict__ out_corr,
                                              float* __restrict__ out_hist,
                                              float* __restrict__ out_yword) {
  // 128 KB: bf16 A fragments [2 subtiles][64 steps][32 lanes][16 elems]
  // during GEMM; reused as f32 [32 pixels][256 K] score tile afterwards.
  __shared__ __align__(32) unsigned short afrag[2 * KSTEPS * 512];

  int b    = blockIdx.x >> 5;                  // image
  int hw0  = (blockIdx.x & 31) * MP;           // 32 contiguous pixels
  int tid  = threadIdx.x;
  int lane = tid & 31;
  int wave = tid >> 5;

  // --- stage normalized A tile (32 pixels x 2048 C) into fragment layout ---
  {
    int m32   = tid & 31;                      // pixel (coalesced 128B over hw)
    int c_off = tid >> 5;                      // 0..15
    int mt    = m32 >> 4;                      // A sub-tile
    int m     = m32 & 15;                      // row within sub-tile
    float invn = x_invn[b * NHW + hw0 + m32];
    const float* xp = x + (size_t)b * NC * NHW + hw0 + m32;
    for (int it = 0; it < NC / 16; ++it) {     // 128 elems per thread
      int c  = it * 16 + c_off;
      int st = c >> 5;
      int kw = c & 31;
      int dl, de;
      frag_pos(m, kw, dl, de);
      afrag[((mt * KSTEPS + st) * 32 + dl) * 16 + de] =
          f2bf(xp[(size_t)c * NHW] * invn);
    }
  }
  __syncthreads();

  // --- GEMM: wave w owns K-tile [w*16, w*16+16); 2 WMMAs per B fragment ----
  v8f acc0 = {0.f, 0.f, 0.f, 0.f, 0.f, 0.f, 0.f, 0.f};
  v8f acc1 = {0.f, 0.f, 0.f, 0.f, 0.f, 0.f, 0.f, 0.f};
  const unsigned short* wf = wfrag + (size_t)wave * (KSTEPS * 512);
  for (int s = 0; s < KSTEPS; ++s) {
    v16bf bb = *reinterpret_cast<const v16bf*>(wf + (s * 32 + lane) * 16);
    v16bf a0 = *reinterpret_cast<const v16bf*>(&afrag[(s * 32 + lane) * 16]);
    v16bf a1 = *reinterpret_cast<const v16bf*>(
        &afrag[((KSTEPS + s) * 32 + lane) * 16]);
    if (s + 1 < KSTEPS)
      __builtin_prefetch((const void*)(wf + ((s + 1) * 32 + lane) * 16), 0, 0);
    acc0 = __builtin_amdgcn_wmma_f32_16x16x32_bf16(
        false, a0, false, bb, (short)0, acc0, false, false);
    acc1 = __builtin_amdgcn_wmma_f32_16x16x32_bf16(
        false, a1, false, bb, (short)0, acc1, false, false);
  }
  __syncthreads();                             // all A reads done; reuse LDS

  float* sc = reinterpret_cast<float*>(afrag); // scores [32 pixels][256 K]
  {
    int mbase = (lane >> 4) << 3;              // lanes 16-31 hold M = r+8
    int n     = wave * 16 + (lane & 15);
#pragma unroll
    for (int r = 0; r < 8; ++r) sc[(mbase + r) * NK + n]        = acc0[r];
#pragma unroll
    for (int r = 0; r < 8; ++r) sc[(16 + mbase + r) * NK + n]   = acc1[r];
  }
  __syncthreads();

  // --- softmax + argmax: wave w handles pixel rows 2w and 2w+1 -------------
  for (int rr = 0; rr < 2; ++rr) {
    int m = wave * 2 + rr;
    float v[8];
    float mx = -INFINITY;
#pragma unroll
    for (int i = 0; i < 8; ++i) {
      v[i] = sc[m * NK + lane + i * 32];
      mx = fmaxf(mx, v[i]);
    }
#pragma unroll
    for (int off = 16; off > 0; off >>= 1) mx = fmaxf(mx, __shfl_xor(mx, off, 32));
    float sum = 0.f;
#pragma unroll
    for (int i = 0; i < 8; ++i) {
      v[i] = __expf(v[i] - mx);
      sum += v[i];
    }
#pragma unroll
    for (int off = 16; off > 0; off >>= 1) sum += __shfl_xor(sum, off, 32);
    float inv = 1.0f / sum;

    // argmax (softmax is monotonic; first-index tie-break like jnp.argmax)
    float bm = -1.0f; int bk = 0;
#pragma unroll
    for (int i = 0; i < 8; ++i) {
      int kk = lane + i * 32;
      if (v[i] > bm || (v[i] == bm && kk < bk)) { bm = v[i]; bk = kk; }
    }
#pragma unroll
    for (int off = 16; off > 0; off >>= 1) {
      float om = __shfl_xor(bm, off, 32);
      int   ok = __shfl_xor(bk, off, 32);
      if (om > bm || (om == bm && ok < bk)) { bm = om; bk = ok; }
    }
#pragma unroll
    for (int i = 0; i < 8; ++i) sc[m * NK + lane + i * 32] = v[i] * inv;
    if (lane == 0) out_yword[b * NK + bk] = 1.0f;  // idempotent store
  }
  __syncthreads();

  // --- histogram partials: one thread per k, one atomic per (b,k,block) ----
  if (tid < NK) {
    float s = 0.f;
#pragma unroll
    for (int m2 = 0; m2 < MP; ++m2) s += sc[m2 * NK + tid];
    atomicAdd(&out_hist[b * NK + tid], s);
  }

  // --- write x_corr [B,K,H,W], coalesced 128B over hw ----------------------
  {
    int m32  = tid & 31;
    int koff = tid >> 5;                       // 0..15
    size_t obase = ((size_t)b * NK) * NHW + hw0 + m32;
    for (int kb = 0; kb < 16; ++kb) {
      int k = kb * 16 + koff;
      out_corr[obase + (size_t)k * NHW] = sc[m32 * NK + k];
    }
  }
}

extern "C" void kernel_launch(void* const* d_in, const int* in_sizes, int n_in,
                              void* d_out, int out_size, void* d_ws, size_t ws_size,
                              hipStream_t stream) {
  const float* x = (const float*)d_in[0];        // [16,2048,32,32]
  const float* w = (const float*)d_in[1];        // [256,2048]

  float* out       = (float*)d_out;
  float* out_corr  = out;                        // 16*256*1024
  float* out_hist  = out + (size_t)NB * NK * NHW;
  float* out_yword = out_hist + NB * NK;

  // workspace: wfrag (1 MB bf16) | w_invn (1 KB) | x_invn (64 KB)
  unsigned short* wfrag = (unsigned short*)d_ws;
  float* w_invn = (float*)((char*)d_ws + (size_t)NTILE * KSTEPS * 512 * 2);
  float* x_invn = w_invn + 256;

  k_zero <<<(2 * NB * NK + 255) / 256, 256, 0, stream>>>(out_hist, 2 * NB * NK);
  k_wnorm<<<NK, 256, 0, stream>>>(w, w_invn);
  k_wfrag<<<(NTILE * KSTEPS * 512) / 256, 256, 0, stream>>>(w, w_invn, wfrag);
  k_xnorm<<<(NB * NHW) / 256, 256, 0, stream>>>(x, x_invn);
  k_main <<<NB * (NHW / MP), 512, 0, stream>>>(x, wfrag, x_invn,
                                               out_corr, out_hist, out_yword);
}
